// Decoder_85590108274851
// MI455X (gfx1250) — matrix operand order repaired and checked by reading, not compiled, not run on hardware
//
#include <hip/hip_runtime.h>
#include <cstdint>
#include <cstddef>

// ---------------------------------------------------------------------------
// MusicVAE hierarchical decoder for MI455X (gfx1250), bf16 WMMA GEMM core.
// 256 sequential LSTM steps; weights converted to bf16 once per launch and
// streamed from L2 (34 MB << 192 MB L2). GEMMs use v_wmma_f32_16x16x32_bf16
// with double-buffered async global->LDS staging (ASYNCcnt) per CDNA5 ISA:
//   wait(tile t) -> barrier -> issue DMA(tile t+1) -> WMMA(tile t)
// so the next tile's DMA is in flight during the current tile's matrix ops.
// ---------------------------------------------------------------------------

typedef __bf16 bf16_t;
typedef __bf16 v16bf __attribute__((ext_vector_type(16)));
typedef __bf16 v8bf  __attribute__((ext_vector_type(8)));
typedef float  v8f   __attribute__((ext_vector_type(8)));
typedef int    v4i_t __attribute__((ext_vector_type(4)));

#define DEV_INLINE __device__ __forceinline__

#define BATCH 256
#define HID   1024
#define ZDIM  512
#define TDIM  512
#define G4H   4096            // 4*HID gate width
#define KCAT  1536            // HID + TDIM (decoder ih input width)
#define LDSROW 40             // 32 bf16 data + 8 pad -> 80B rows (16B aligned)

// ---- gfx1250 async global->LDS staging -----------------------------------
DEV_INLINE void async_b128(const bf16_t* gsrc, bf16_t* lds_dst) {
#if __has_builtin(__builtin_amdgcn_global_load_async_to_lds_b128)
  __builtin_amdgcn_global_load_async_to_lds_b128(
      (__attribute__((address_space(1))) v4i_t*)gsrc,
      (__attribute__((address_space(3))) v4i_t*)lds_dst, 0, 0);
#else
  *(uint4*)lds_dst = *(const uint4*)gsrc;   // synchronous fallback
#endif
}

DEV_INLINE void wait_async_all() {
#if __has_builtin(__builtin_amdgcn_s_wait_asynccnt)
  __builtin_amdgcn_s_wait_asynccnt(0);
#else
  asm volatile("s_wait_asynccnt 0" ::: "memory");
#endif
}

DEV_INLINE float sigmoidf_fast(float x) { return 1.0f / (1.0f + __expf(-x)); }

// issue one 64x32 A tile + 128-row B (=W^T 32x128) tile into an LDS buffer
DEV_INLINE void issue_tile(const bf16_t* A, int lda, const bf16_t* W, int K,
                           int k0, int mtile, int ntile, int lrow, int lch,
                           bf16_t* AshBuf, bf16_t* BshBuf)
{
  async_b128(A + (size_t)(mtile + lrow) * lda + k0 + lch,
             &AshBuf[lrow * LDSROW + lch]);
  async_b128(W + (size_t)(ntile + lrow) * K + k0 + lch,
             &BshBuf[lrow * LDSROW + lch]);
  async_b128(W + (size_t)(ntile + 64 + lrow) * K + k0 + lch,
             &BshBuf[(64 + lrow) * LDSROW + lch]);
}

// ---------------------------------------------------------------------------
// GEMM: out[m,n] = sum_k A1[m,k]*W1[n,k] (k<K1)  +  sum_k A2[m,k]*W2[n,k] (k<K2)
//                  + bias[n]
// EPI==0: plain fp32 store (LSTM gates).  EPI==1: sigmoid, store fp32 to outp
//         (row stride ldo) AND bf16 into xnote[m*KCAT + HID + n].
// Grid: x = N/128 tiles, y = M/64 tiles. Block = 256 threads = 8 waves,
// wave tile 32x32 (2x2 WMMA 16x16x32 bf16 fragments, fp32 accum).
// ---------------------------------------------------------------------------
template <int EPI>
__global__ __launch_bounds__(256)
void gemm_wmma_kernel(const bf16_t* __restrict__ A1, int lda1,
                      const bf16_t* __restrict__ W1, int K1,
                      const bf16_t* __restrict__ A2, int lda2,
                      const bf16_t* __restrict__ W2, int K2,
                      const float*  __restrict__ bias,
                      float*        __restrict__ outp, long ldo,
                      bf16_t*       __restrict__ xnote)
{
  __shared__ __align__(16) bf16_t Ash[2][64 * LDSROW];
  __shared__ __align__(16) bf16_t Bsh[2][128 * LDSROW];

  const int tid   = threadIdx.x;
  const int lane  = tid & 31;
  const int wave  = tid >> 5;
  const int waveM = (wave >> 2) << 5;   // 0 or 32
  const int waveN = (wave & 3) << 5;    // 0,32,64,96
  const int g     = lane >> 4;          // half-wave group
  const int ml    = lane & 15;

  const int mtile = blockIdx.y << 6;    // 64-row M tile
  const int ntile = blockIdx.x << 7;    // 128-col N tile

  const int lrow = tid >> 2;            // 0..63
  const int lch  = (tid & 3) << 3;      // 0,8,16,24 bf16 elems (16B chunks)

  const int steps1 = K1 >> 5;
  const int steps2 = K2 >> 5;
  const int total  = steps1 + steps2;

  v8f acc[2][2] = {};

  // prologue: DMA tile 0 into buffer 0 (K1 >= 512 always, so total >= 1)
  issue_tile(A1, lda1, W1, K1, 0, mtile, ntile, lrow, lch, Ash[0], Bsh[0]);

  for (int t = 0; t < total; ++t) {
    wait_async_all();       // this wave's tile-t chunks have landed in LDS
    __syncthreads();        // all waves' tile-t data in; all done reading buf^1

    if (t + 1 < total) {    // issue DMA for tile t+1 into the other buffer
      const int tt = t + 1;
      const bool p1 = tt < steps1;
      issue_tile(p1 ? A1 : A2, p1 ? lda1 : lda2,
                 p1 ? W1 : W2, p1 ? K1 : K2,
                 (p1 ? tt : tt - steps1) << 5,
                 mtile, ntile, lrow, lch, Ash[(t + 1) & 1], Bsh[(t + 1) & 1]);
    }

    const bf16_t* Asb = Ash[t & 1];
    const bf16_t* Bsb = Bsh[t & 1];

    // A fragment (16x32 bf16): lane holds row m = ml, K = 8g+0..7, 16+8g+0..7
    v16bf afr[2], bfr[2];
#pragma unroll
    for (int mt = 0; mt < 2; ++mt) {
      const bf16_t* base = &Asb[(waveM + (mt << 4) + ml) * LDSROW];
      v8bf lo = *(const v8bf*)(base + (g << 3));
      v8bf hi = *(const v8bf*)(base + 16 + (g << 3));
      afr[mt] = __builtin_shufflevector(lo, hi, 0,1,2,3,4,5,6,7,
                                                8,9,10,11,12,13,14,15);
    }
    // B fragment (32x16 bf16): lane holds col n = ml, K = 8g+0..7, 16+8g+0..7
#pragma unroll
    for (int nt = 0; nt < 2; ++nt) {
      const bf16_t* base = &Bsb[(waveN + (nt << 4) + ml) * LDSROW];
      v8bf lo = *(const v8bf*)(base + (g << 3));
      v8bf hi = *(const v8bf*)(base + 16 + (g << 3));
      bfr[nt] = __builtin_shufflevector(lo, hi, 0,1,2,3,4,5,6,7,
                                                8,9,10,11,12,13,14,15);
    }
#pragma unroll
    for (int mt = 0; mt < 2; ++mt)
#pragma unroll
      for (int nt = 0; nt < 2; ++nt)
        acc[mt][nt] = __builtin_amdgcn_wmma_f32_16x16x32_bf16(
            false, afr[mt], false, bfr[nt], (short)0, acc[mt][nt],
            false, false);
  }

  // Epilogue. C/D layout: VGPR r -> lanes0-15: M=r, lanes16-31: M=8+r; N=lane&15
#pragma unroll
  for (int mt = 0; mt < 2; ++mt) {
#pragma unroll
    for (int nt = 0; nt < 2; ++nt) {
      const int   n  = ntile + waveN + (nt << 4) + ml;
      const float bv = bias[n];
#pragma unroll
      for (int r = 0; r < 8; ++r) {
        const int m = mtile + waveM + (mt << 4) + (g << 3) + r;
        float v = acc[mt][nt][r] + bv;
        if (EPI == 1) {
          v = sigmoidf_fast(v);
          outp[(size_t)m * ldo + n] = v;
          xnote[(size_t)m * KCAT + HID + n] = (bf16_t)v;
        } else {
          outp[(size_t)m * ldo + n] = v;
        }
      }
    }
  }
}

// ---- elementwise LSTM cell: gate order i,f,g,o ----------------------------
__global__ __launch_bounds__(256)
void lstm_cell_kernel(const float* __restrict__ gates,
                      float* __restrict__ c,
                      bf16_t* __restrict__ hout, int ldh)
{
  const int i = blockIdx.x * 256 + threadIdx.x;  // 0..B*H-1
  const int b = i >> 10, h = i & (HID - 1);
  const float* gr = gates + (size_t)b * G4H;
  const float ig = sigmoidf_fast(gr[h]);
  const float fg = sigmoidf_fast(gr[HID + h]);
  const float gg = tanhf(gr[2 * HID + h]);
  const float og = sigmoidf_fast(gr[3 * HID + h]);
  const float cn = fg * c[i] + ig * gg;
  c[i] = cn;
  hout[(size_t)b * ldh + h] = (bf16_t)(og * tanhf(cn));
}

__global__ __launch_bounds__(256)
void cvt_f32_bf16_kernel(const float* __restrict__ s, bf16_t* __restrict__ d, int n)
{
  const int i = blockIdx.x * 256 + threadIdx.x;
  if (i < n) d[i] = (bf16_t)s[i];
}

__global__ __launch_bounds__(256)
void bias_sum_kernel(const float* __restrict__ a, const float* __restrict__ b,
                     float* __restrict__ o)
{
  const int i = blockIdx.x * 256 + threadIdx.x;
  o[i] = a[i] + b[i];
}

// z[:, 16*level, 0, :] -> bf16 [B, Z]
__global__ __launch_bounds__(256)
void load_z_kernel(const float* __restrict__ z, bf16_t* __restrict__ xz, int level)
{
  const int i = blockIdx.x * 256 + threadIdx.x;  // 0..B*Z-1
  const int b = i >> 9, k = i & (ZDIM - 1);
  xz[i] = (bf16_t)z[(size_t)b * (256 * ZDIM) + (size_t)(level * 16) * ZDIM + k];
}

// conductor h0 = 0 (lives in X[:, :HID]), conductor c0 = 0
__global__ __launch_bounds__(256)
void init_start_kernel(float* __restrict__ Cc, bf16_t* __restrict__ X)
{
  const int i = blockIdx.x * 256 + threadIdx.x;  // 0..B*H-1
  const int b = i >> 10, h = i & (HID - 1);
  Cc[i] = 0.0f;
  X[(size_t)b * KCAT + h] = (bf16_t)0.0f;
}

// per level: decoder state <- dec_h0/dec_c0[l], note0 = 0
__global__ __launch_bounds__(256)
void init_level_kernel(const float* __restrict__ h0, const float* __restrict__ c0,
                       float* __restrict__ Cd, bf16_t* __restrict__ Hd,
                       bf16_t* __restrict__ X)
{
  const int i = blockIdx.x * 256 + threadIdx.x;  // 0..B*H-1
  const int b = i >> 10, h = i & (HID - 1);
  Cd[i] = c0[i];
  Hd[i] = (bf16_t)h0[i];
  if (h < TDIM) X[(size_t)b * KCAT + HID + h] = (bf16_t)0.0f;
}

// ---------------------------------------------------------------------------
extern "C" void kernel_launch(void* const* d_in, const int* in_sizes, int n_in,
                              void* d_out, int out_size, void* d_ws, size_t ws_size,
                              hipStream_t stream)
{
  (void)in_sizes; (void)n_in; (void)out_size; (void)ws_size;

  const float* z        = (const float*)d_in[0];
  const float* dec_h0   = (const float*)d_in[1];
  const float* dec_c0   = (const float*)d_in[2];
  // d_in[3], d_in[4]: linear_in (unused by the reference forward)
  const float* cond_Wih = (const float*)d_in[5];
  const float* cond_Whh = (const float*)d_in[6];
  const float* cond_bih = (const float*)d_in[7];
  const float* cond_bhh = (const float*)d_in[8];
  const float* dec_Wih  = (const float*)d_in[9];
  const float* dec_Whh  = (const float*)d_in[10];
  const float* dec_bih  = (const float*)d_in[11];
  const float* dec_bhh  = (const float*)d_in[12];
  const float* out_W    = (const float*)d_in[13];
  const float* out_b    = (const float*)d_in[14];
  float* dout = (float*)d_out;

  char* p = (char*)d_ws;
  auto take = [&](size_t bytes) -> char* {
    char* r = p; p += (bytes + 255) & ~(size_t)255; return r;
  };
  bf16_t* Wc_ih = (bf16_t*)take((size_t)G4H * ZDIM * 2);
  bf16_t* Wc_hh = (bf16_t*)take((size_t)G4H * HID  * 2);
  bf16_t* Wd_ih = (bf16_t*)take((size_t)G4H * KCAT * 2);
  bf16_t* Wd_hh = (bf16_t*)take((size_t)G4H * HID  * 2);
  bf16_t* Wo    = (bf16_t*)take((size_t)TDIM * HID * 2);
  float*  cbias = (float*) take((size_t)G4H * 4);
  float*  dbias = (float*) take((size_t)G4H * 4);
  bf16_t* X     = (bf16_t*)take((size_t)BATCH * KCAT * 2);  // [emb | note] bf16
  bf16_t* Xz    = (bf16_t*)take((size_t)BATCH * ZDIM * 2);
  bf16_t* Hd    = (bf16_t*)take((size_t)BATCH * HID * 2);
  float*  Cc    = (float*) take((size_t)BATCH * HID * 4);
  float*  Cd    = (float*) take((size_t)BATCH * HID * 4);
  float*  gates = (float*) take((size_t)BATCH * G4H * 4);

  auto cvt = [&](const float* s, bf16_t* d, int n) {
    cvt_f32_bf16_kernel<<<(n + 255) / 256, 256, 0, stream>>>(s, d, n);
  };
  cvt(cond_Wih, Wc_ih, G4H * ZDIM);
  cvt(cond_Whh, Wc_hh, G4H * HID);
  cvt(dec_Wih,  Wd_ih, G4H * KCAT);
  cvt(dec_Whh,  Wd_hh, G4H * HID);
  cvt(out_W,    Wo,    TDIM * HID);
  bias_sum_kernel<<<G4H / 256, 256, 0, stream>>>(cond_bih, cond_bhh, cbias);
  bias_sum_kernel<<<G4H / 256, 256, 0, stream>>>(dec_bih,  dec_bhh,  dbias);
  init_start_kernel<<<BATCH * HID / 256, 256, 0, stream>>>(Cc, X);

  const dim3 gGates(G4H / 128, BATCH / 64);   // 32 x 4 workgroups
  const dim3 gOut(TDIM / 128, BATCH / 64);    //  4 x 4 workgroups
  const int  ewBlocks = BATCH * HID / 256;

  for (int l = 0; l < 16; ++l) {
    // conductor step: gates = z_l @ Wih^T + h_c @ Whh^T + b ; cell
    load_z_kernel<<<BATCH * ZDIM / 256, 256, 0, stream>>>(z, Xz, l);
    gemm_wmma_kernel<0><<<gGates, 256, 0, stream>>>(
        Xz, ZDIM, Wc_ih, ZDIM, X, KCAT, Wc_hh, HID,
        cbias, gates, (long)G4H, nullptr);
    lstm_cell_kernel<<<ewBlocks, 256, 0, stream>>>(gates, Cc, X, KCAT); // h_c -> emb

    init_level_kernel<<<ewBlocks, 256, 0, stream>>>(
        dec_h0 + (size_t)l * BATCH * HID, dec_c0 + (size_t)l * BATCH * HID,
        Cd, Hd, X);

    for (int n = 0; n < 16; ++n) {
      gemm_wmma_kernel<0><<<gGates, 256, 0, stream>>>(
          X, KCAT, Wd_ih, KCAT, Hd, HID, Wd_hh, HID,
          dbias, gates, (long)G4H, nullptr);
      lstm_cell_kernel<<<ewBlocks, 256, 0, stream>>>(gates, Cd, Hd, HID);

      const int step = l * 16 + n;
      gemm_wmma_kernel<1><<<gOut, 256, 0, stream>>>(
          Hd, HID, Wo, HID, Hd, HID, Wo, /*K2=*/0,
          out_b, dout + (size_t)step * TDIM, (long)(256 * TDIM), X);
    }
  }
}
